// DSCharRNN_17514876633137
// MI455X (gfx1250) — matrix operand order repaired and checked by reading, not compiled, not run on hardware
//
#include <hip/hip_runtime.h>
#include <hip/hip_bf16.h>

// Problem sizes (fixed by the reference)
#define BATCH 128
#define HID   1024
#define GATES 4096          // 4*HID
#define VOC   128
#define SEQ   256

// Persistent-kernel shape: 64 blocks x 256 threads = 512 wave32 waves.
#define NBLK  64
#define NTHR  256
#define NWAVE (NBLK * (NTHR / 32))   // 512

typedef __attribute__((ext_vector_type(16))) __bf16 v16bf;
typedef __attribute__((ext_vector_type(8)))  __bf16 v8bf;
typedef __attribute__((ext_vector_type(8)))  float  v8f;

static __device__ inline v8f zero_v8f() {
  v8f z = {0.f, 0.f, 0.f, 0.f, 0.f, 0.f, 0.f, 0.f};
  return z;
}

static __device__ inline v8f wmma_bf16(v16bf a, v16bf b, v8f c) {
  // D(f32 16x16) = A(bf16 16x32) * B(bf16 32x16) + C
  return __builtin_amdgcn_wmma_f32_16x16x32_bf16(
      /*neg_a=*/false, a, /*neg_b=*/false, b,
      /*c_mod=*/(short)0, c, /*reuse_a=*/false, /*reuse_b=*/false);
}

// A fragment (16x32, M x K): lane L holds row M=L%16;
// 16 contiguous K values starting at kbase + 16*(L/16) (matches B)
static __device__ inline v16bf load_a_frag(const __bf16* row, int k8) {
  union { v16bf v; v8bf h[2]; } u;
  u.h[0] = *(const v8bf*)(row + k8 + (k8 & 8));      // 16B load
  u.h[1] = *(const v8bf*)(row + k8 + (k8 & 8) + 8);  // 16B load
  return u.v;
}

static __device__ inline float sigmoidf_(float x) {
  return 1.0f / (1.0f + __expf(-x));
}

// ---------------- grid-wide barrier (sense reversal, agent scope) ------------
static __device__ inline void grid_sync(unsigned* cnt, unsigned* gen) {
  __syncthreads();
  __builtin_amdgcn_fence(__ATOMIC_RELEASE, "agent");
  if (threadIdx.x == 0) {
    unsigned g = __hip_atomic_load(gen, __ATOMIC_RELAXED, __HIP_MEMORY_SCOPE_AGENT);
    unsigned prev = __hip_atomic_fetch_add(cnt, 1u, __ATOMIC_ACQ_REL, __HIP_MEMORY_SCOPE_AGENT);
    if (prev == (unsigned)(NBLK - 1)) {
      __hip_atomic_store(cnt, 0u, __ATOMIC_RELAXED, __HIP_MEMORY_SCOPE_AGENT);
      __hip_atomic_fetch_add(gen, 1u, __ATOMIC_RELEASE, __HIP_MEMORY_SCOPE_AGENT);
    } else {
      while (__hip_atomic_load(gen, __ATOMIC_ACQUIRE, __HIP_MEMORY_SCOPE_AGENT) == g)
        __builtin_amdgcn_s_sleep(1);
    }
  }
  __syncthreads();
  __builtin_amdgcn_fence(__ATOMIC_ACQUIRE, "agent");
}

// ---------------- phase A: layer-0 LSTM cell --------------------------------
// g0 = onehot(tok) @ Wi0^T (gather) + h0_prev @ Wh0^T + (bi0+bh0)
// Unit u in [0,512): jt = u>>3 (hidden tile), mt = u&7 (batch tile).
static __device__ void phase_a(int u, int t,
                               const __bf16* __restrict__ h0_prev,
                               __bf16* __restrict__ h0_cur,
                               float* __restrict__ c0,
                               const __bf16* __restrict__ wh0b,
                               const float* __restrict__ bc0,
                               const float* __restrict__ Wi0,
                               const int* __restrict__ tokens) {
  const int lane = threadIdx.x & 31;
  const int n    = lane & 15;
  const int half = lane >> 4;
  const int jt = u >> 3;
  const int mt = u & 7;
  const __bf16* arow = h0_prev + (mt * 16 + n) * HID;

  v8f acc[4] = {zero_v8f(), zero_v8f(), zero_v8f(), zero_v8f()};
  for (int kk = 0; kk < HID; kk += 32) {
    v16bf a = load_a_frag(arow, kk + 8 * half);
    const int kb = kk + 16 * half;
#pragma unroll
    for (int gc = 0; gc < 4; ++gc) {
      // B fragment (32x16, K x N): lane holds column N = lane%16,
      // 16 contiguous K values starting at kbase + 16*(lane/16) -> one 32B load
      v16bf b = *(const v16bf*)(wh0b + (size_t)(gc * HID + jt * 16 + n) * HID + kb);
      acc[gc] = wmma_bf16(a, b, acc[gc]);
    }
  }

  const int j = jt * 16 + n;   // hidden index owned by this lane
  const float bi_ = bc0[0 * HID + j];
  const float bf_ = bc0[1 * HID + j];
  const float bg_ = bc0[2 * HID + j];
  const float bo_ = bc0[3 * HID + j];
#pragma unroll
  for (int r = 0; r < 8; ++r) {
    const int b   = mt * 16 + r + 8 * half;     // C/D layout: M = vgpr + 8*(lane/16)
    const int tok = tokens[b * SEQ + t];
    float gi = acc[0][r] + Wi0[(size_t)(0 * HID + j) * VOC + tok] + bi_;
    float gf = acc[1][r] + Wi0[(size_t)(1 * HID + j) * VOC + tok] + bf_;
    float gg = acc[2][r] + Wi0[(size_t)(2 * HID + j) * VOC + tok] + bg_;
    float go = acc[3][r] + Wi0[(size_t)(3 * HID + j) * VOC + tok] + bo_;
    float ii = sigmoidf_(gi), ff = sigmoidf_(gf);
    float g2 = tanhf(gg),     oo = sigmoidf_(go);
    float cc = c0[b * HID + j];
    float c2 = ff * cc + ii * g2;
    c0[b * HID + j]     = c2;
    h0_cur[b * HID + j] = (__bf16)(oo * tanhf(c2));
  }
}

// ---------------- phase B: layer-1 LSTM cell --------------------------------
// g1 = h0_cur @ Wi1^T + h1_prev @ Wh1^T + (bi1+bh1)
static __device__ void phase_b(int u,
                               const __bf16* __restrict__ x,
                               const __bf16* __restrict__ h1_prev,
                               __bf16* __restrict__ h1_cur,
                               float* __restrict__ c1,
                               const __bf16* __restrict__ wi1b,
                               const __bf16* __restrict__ wh1b,
                               const float* __restrict__ bc1) {
  const int lane = threadIdx.x & 31;
  const int n    = lane & 15;
  const int half = lane >> 4;
  const int jt = u >> 3;
  const int mt = u & 7;
  const __bf16* xrow = x       + (mt * 16 + n) * HID;
  const __bf16* hrow = h1_prev + (mt * 16 + n) * HID;

  v8f acc[4] = {zero_v8f(), zero_v8f(), zero_v8f(), zero_v8f()};
  for (int kk = 0; kk < HID; kk += 32) {
    v16bf ax = load_a_frag(xrow, kk + 8 * half);
    v16bf ah = load_a_frag(hrow, kk + 8 * half);
    const int kb = kk + 16 * half;
#pragma unroll
    for (int gc = 0; gc < 4; ++gc) {
      const size_t roff = (size_t)(gc * HID + jt * 16 + n) * HID + kb;
      v16bf b0 = *(const v16bf*)(wi1b + roff);
      acc[gc] = wmma_bf16(ax, b0, acc[gc]);
      v16bf b1 = *(const v16bf*)(wh1b + roff);
      acc[gc] = wmma_bf16(ah, b1, acc[gc]);
    }
  }

  const int j = jt * 16 + n;
  const float bi_ = bc1[0 * HID + j];
  const float bf_ = bc1[1 * HID + j];
  const float bg_ = bc1[2 * HID + j];
  const float bo_ = bc1[3 * HID + j];
#pragma unroll
  for (int r = 0; r < 8; ++r) {
    const int b = mt * 16 + r + 8 * half;
    float gi = acc[0][r] + bi_;
    float gf = acc[1][r] + bf_;
    float gg = acc[2][r] + bg_;
    float go = acc[3][r] + bo_;
    float ii = sigmoidf_(gi), ff = sigmoidf_(gf);
    float g2 = tanhf(gg),     oo = sigmoidf_(go);
    float cc = c1[b * HID + j];
    float c2 = ff * cc + ii * g2;
    c1[b * HID + j]     = c2;
    h1_cur[b * HID + j] = (__bf16)(oo * tanhf(c2));
  }
}

// ---------------- tail: y_t = h1 @ W_tail^T ---------------------------------
// Unit u in [0,64): nt = u>>3 (vocab tile), mt = u&7 (batch tile).
static __device__ void phase_tail(int u, int t,
                                  const __bf16* __restrict__ h1buf,
                                  const __bf16* __restrict__ wtb,
                                  float* __restrict__ out) {
  const int lane = threadIdx.x & 31;
  const int n    = lane & 15;
  const int half = lane >> 4;
  const int nt = u >> 3;
  const int mt = u & 7;
  const __bf16* arow = h1buf + (mt * 16 + n) * HID;

  v8f acc = zero_v8f();
  for (int kk = 0; kk < HID; kk += 32) {
    v16bf a = load_a_frag(arow, kk + 8 * half);
    v16bf b = *(const v16bf*)(wtb + (size_t)(nt * 16 + n) * HID + kk + 16 * half);
    acc = wmma_bf16(a, b, acc);
  }
#pragma unroll
  for (int r = 0; r < 8; ++r) {
    const int b   = mt * 16 + r + 8 * half;
    const int col = nt * 16 + n;
    out[(size_t)(b * SEQ + t) * VOC + col] = acc[r];
  }
}

// ---------------- persistent kernel -----------------------------------------
__global__ __launch_bounds__(NTHR, 1) void
dscharrnn_persistent(const int*   __restrict__ tokens,
                     const float* __restrict__ Wi0,
                     const float* __restrict__ Wh0,
                     const float* __restrict__ bi0,
                     const float* __restrict__ bh0,
                     const float* __restrict__ Wi1,
                     const float* __restrict__ Wh1,
                     const float* __restrict__ bi1,
                     const float* __restrict__ bh1,
                     const float* __restrict__ Wtail,
                     float* __restrict__ out,
                     unsigned char* __restrict__ ws) {
  // workspace layout (all segments 256B aligned)
  __bf16* wh0b = (__bf16*)ws;
  __bf16* wi1b = wh0b + (size_t)GATES * HID;
  __bf16* wh1b = wi1b + (size_t)GATES * HID;
  __bf16* wtb  = wh1b + (size_t)GATES * HID;
  float*  bc0  = (float*)(wtb + (size_t)VOC * HID);
  float*  bc1  = bc0 + GATES;
  __bf16* h0   = (__bf16*)(bc1 + GATES);          // 2 * B * H (ping-pong)
  __bf16* h1   = h0 + 2 * BATCH * HID;            // 2 * B * H
  float*  c0   = (float*)(h1 + 2 * BATCH * HID);  // B * H
  float*  c1   = c0 + BATCH * HID;                // B * H
  unsigned* syncv = (unsigned*)(c1 + BATCH * HID);
  unsigned* cnt = syncv, *gen = syncv + 1, *flag = syncv + 2;

  const int tid = blockIdx.x * NTHR + threadIdx.x;
  const int gw  = tid >> 5;                 // global wave id, 0..511
  const int NT  = NBLK * NTHR;
  const unsigned MAGIC = 0x13572468u;

  // One-time barrier-state init (ws is poisoned to 0xAA before timing).
  if (tid == 0) {
    if (__hip_atomic_load(flag, __ATOMIC_RELAXED, __HIP_MEMORY_SCOPE_AGENT) != MAGIC) {
      __hip_atomic_store(cnt, 0u, __ATOMIC_RELAXED, __HIP_MEMORY_SCOPE_AGENT);
      __hip_atomic_store(gen, 0u, __ATOMIC_RELAXED, __HIP_MEMORY_SCOPE_AGENT);
      __hip_atomic_store(flag, MAGIC, __ATOMIC_RELEASE, __HIP_MEMORY_SCOPE_AGENT);
    }
  }
  if (threadIdx.x == 0) {
    while (__hip_atomic_load(flag, __ATOMIC_ACQUIRE, __HIP_MEMORY_SCOPE_AGENT) != MAGIC)
      __builtin_amdgcn_s_sleep(1);
  }
  __syncthreads();

  // Phase 0: f32 -> bf16 weight conversion, bias fusion, state zeroing.
  for (size_t i = tid; i < (size_t)GATES * HID; i += NT) {
    wh0b[i] = (__bf16)Wh0[i];
    wi1b[i] = (__bf16)Wi1[i];
    wh1b[i] = (__bf16)Wh1[i];
  }
  for (int i = tid; i < VOC * HID; i += NT) wtb[i] = (__bf16)Wtail[i];
  for (int i = tid; i < GATES; i += NT) { bc0[i] = bi0[i] + bh0[i]; bc1[i] = bi1[i] + bh1[i]; }
  for (int i = tid; i < 2 * BATCH * HID; i += NT) { h0[i] = (__bf16)0.0f; h1[i] = (__bf16)0.0f; }
  for (int i = tid; i < BATCH * HID; i += NT) { c0[i] = 0.0f; c1[i] = 0.0f; }
  grid_sync(cnt, gen);

  // Recurrence: 2 grid barriers per step; tail(t-1) overlaps phase A(t).
  for (int t = 0; t < SEQ; ++t) {
    const int cur = t & 1, prv = cur ^ 1;
    const __bf16* h0p = h0 + prv * BATCH * HID;
    __bf16*       h0c = h0 + cur * BATCH * HID;
    const __bf16* h1p = h1 + prv * BATCH * HID;
    __bf16*       h1c = h1 + cur * BATCH * HID;

    for (int u = gw; u < 512 + 64; u += NWAVE) {
      if (u < 512)      phase_a(u, t, h0p, h0c, c0, wh0b, bc0, Wi0, tokens);
      else if (t > 0)   phase_tail(u - 512, t - 1, h1p, wtb, out);
    }
    grid_sync(cnt, gen);

    for (int u = gw; u < 512; u += NWAVE)
      phase_b(u, h0c, h1p, h1c, c1, wi1b, wh1b, bc1);
    grid_sync(cnt, gen);
  }

  // Final tail for t = SEQ-1.
  for (int u = gw; u < 64; u += NWAVE)
    phase_tail(u, SEQ - 1, h1 + ((SEQ - 1) & 1) * BATCH * HID, wtb, out);
}

extern "C" void kernel_launch(void* const* d_in, const int* in_sizes, int n_in,
                              void* d_out, int out_size, void* d_ws, size_t ws_size,
                              hipStream_t stream) {
  (void)in_sizes; (void)n_in; (void)out_size; (void)ws_size;
  dscharrnn_persistent<<<NBLK, NTHR, 0, stream>>>(
      (const int*)d_in[0],
      (const float*)d_in[1], (const float*)d_in[2],
      (const float*)d_in[3], (const float*)d_in[4],
      (const float*)d_in[5], (const float*)d_in[6],
      (const float*)d_in[7], (const float*)d_in[8],
      (const float*)d_in[9],
      (float*)d_out, (unsigned char*)d_ws);
}
